// AutoEncoder_4741643894842
// MI455X (gfx1250) — compile-verified, matrix operand order checked by reading, not run on hardware
//
#include <hip/hip_runtime.h>
#include <hip/hip_bf16.h>

typedef _Float16 f16_t;
typedef __attribute__((ext_vector_type(16))) _Float16 v16h;
typedef __attribute__((ext_vector_type(8)))  _Float16 v8h;
typedef __attribute__((ext_vector_type(8)))  float    v8f;
typedef unsigned int u32x4 __attribute__((ext_vector_type(4)));
typedef int          i32x4 __attribute__((ext_vector_type(4)));
typedef int          i32x8 __attribute__((ext_vector_type(8)));

static constexpr int Hc = 192, Wc = 192, Bc = 2;
static constexpr int HWc  = Hc * Wc;        // 36864
static constexpr int PIXc = Bc * HWc;       // 73728
static constexpr int Ec   = 20;
static constexpr int MTc  = (Wc / 16) * Hc * Bc;  // 4608 M-tiles (16 px along W)

// ---------------------------------------------------------------------------
// Convert x (f32 NCHW, C=3) -> f16 NHWC (C=3)
// ---------------------------------------------------------------------------
__global__ __launch_bounds__(256) void k_x_to_f16(const float* __restrict__ x,
                                                  f16_t* __restrict__ out) {
  int i = blockIdx.x * blockDim.x + threadIdx.x;
  if (i >= Bc * 3 * HWc) return;
  int w = i % Wc;
  int h = (i / Wc) % Hc;
  int c = (i / HWc) % 3;
  int b = i / (3 * HWc);
  out[((size_t)((b * Hc + h) * Wc + w)) * 3 + c] = (f16_t)x[i];
}

// ---------------------------------------------------------------------------
// Fold conv-bias (+ optional BN) into per-channel alpha/beta
// ---------------------------------------------------------------------------
__global__ __launch_bounds__(256) void k_affine(const float* bias, const float* g,
                                                const float* bb, const float* mn,
                                                const float* vr, float* alpha,
                                                float* beta, int C, int bn) {
  int c = blockIdx.x * blockDim.x + threadIdx.x;
  if (c >= C) return;
  float bi = bias ? bias[c] : 0.0f;
  if (bn) {
    float s = g[c] * rsqrtf(vr[c] + 1e-5f);
    alpha[c] = s;
    beta[c]  = bi * s + bb[c] - mn[c] * s;
  } else {
    alpha[c] = 1.0f;
    beta[c]  = bi;
  }
}

// ---------------------------------------------------------------------------
// Pack f32 OIHW weights into the WMMA 16-bit B-fragment layout:
//   packed[((nt*KB + kk)*32 + lane)*16 + j]
// ---------------------------------------------------------------------------
__global__ __launch_bounds__(256) void k_pack(const float* __restrict__ w,
                                              f16_t* __restrict__ packed,
                                              int Cout, int Cin, int taps) {
  int K  = (Cin == 3) ? 32 : taps * Cin;
  int KB = K / 32;
  int NT = Cout / 16;
  int total = NT * KB * 512;
  int i = blockIdx.x * blockDim.x + threadIdx.x;
  if (i >= total) return;
  int j    = i & 15;
  int lane = (i >> 4) & 31;
  int kk   = (i >> 9) % KB;
  int nt   = i / (512 * KB);
  int n    = nt * 16 + (lane & 15);
  int kb   = (lane < 16) ? 0 : 8;
  int k    = kk * 32 + ((j < 8) ? (kb + j) : (16 + kb + (j - 8)));
  float val = 0.0f;
  if (n < Cout) {
    if (Cin == 3) {
      if (k < 27) {
        int s = k / 3, c = k % 3;
        val = w[(n * 3 + c) * 9 + s];
      }
    } else {
      int s = k / Cin, c = k % Cin;
      if (s < taps) val = w[((size_t)n * Cin + c) * taps + s];
    }
  }
  packed[i] = (f16_t)val;
}

// ---------------------------------------------------------------------------
// 3x3 conv, Cin=3 (K padded 27->32, single WMMA per tile). in: f16 NHWC C=3.
// ---------------------------------------------------------------------------
template <int COUT, bool LRELU>
__global__ __launch_bounds__(128) void k_conv_c3(const f16_t* __restrict__ in,
                                                 const f16_t* __restrict__ wp,
                                                 const float* __restrict__ alpha,
                                                 const float* __restrict__ beta,
                                                 f16_t* __restrict__ out) {
  constexpr int NT = COUT / 16;
  int wave = blockIdx.x * 4 + (threadIdx.x >> 5);
  int lane = threadIdx.x & 31;
  int nt = wave % NT, mt = wave / NT;
  if (mt >= MTc) return;
  int w0 = (mt % (Wc / 16)) * 16;
  int h  = (mt / (Wc / 16)) % Hc;
  int b  = mt / ((Wc / 16) * Hc);
  int m    = lane & 15;
  int kb   = (lane < 16) ? 0 : 8;
  int wpix = w0 + m;

  v16h a = {};
#pragma unroll
  for (int j = 0; j < 16; ++j) {
    int k = (j < 8) ? (kb + j) : (16 + kb + (j - 8));
    if (k < 27) {
      int s = k / 3, c = k % 3;
      int hh = h + s / 3 - 1, ww = wpix + s % 3 - 1;
      if (hh >= 0 && hh < Hc && ww >= 0 && ww < Wc)
        a[j] = in[((size_t)((b * Hc + hh) * Wc + ww)) * 3 + c];
    }
  }
  v16h bf = *(const v16h*)(wp + ((size_t)(nt * 32 + lane)) * 16);
  v8f acc = {};
  acc = __builtin_amdgcn_wmma_f32_16x16x32_f16(false, a, false, bf, (short)0, acc,
                                               false, false);
  int n = nt * 16 + m;
  float al = alpha[n], be = beta[n];
  int mb = (lane < 16) ? 0 : 8;
#pragma unroll
  for (int r = 0; r < 8; ++r) {
    float v = acc[r] * al + be;
    if (LRELU) v = (v >= 0.0f) ? v : 0.2f * v;
    out[((size_t)((b * Hc + h) * Wc + w0 + mb + r)) * COUT + n] = (f16_t)v;
  }
}

// ---------------------------------------------------------------------------
// Generic WMMA conv: TDM-staged weights in LDS, 2 M-tiles x NW N-tiles / wave.
//   - block = 4 waves sharing one group of NW*16 output channels;
//     wave i handles M-tiles (8*blockGroup + 2i, +1).
//   - wave 0 issues one tensor_load_to_lds covering the contiguous packed-B
//     group, waits on TENSORcnt, then the block barriers.
//   - K-loop: 2 A-fragments (global b128 x2 each), NW B-frags (ds_load_b128 x2),
//     2*NW WMMAs -> A/B traffic amortized over 8 matrix ops.
// KS=3 ('SAME' 3x3) or KS=1 (1x1 GEMM). MASK: write only where idx[pix]==expert.
// ---------------------------------------------------------------------------
template <int CIN, int COUT, int KS, int NW, bool LRELU, bool MASK>
__global__ __launch_bounds__(128) void k_conv_wmma(const f16_t* __restrict__ in,
                                                   const f16_t* __restrict__ wp,
                                                   const float* __restrict__ alpha,
                                                   const float* __restrict__ beta,
                                                   f16_t* __restrict__ out,
                                                   const int* __restrict__ idx,
                                                   int expert) {
  constexpr int NT = COUT / 16;
  constexpr int NG = NT / NW;           // channel groups
  constexpr int CB = CIN / 32;
  constexpr int KB = KS * KS * CB;      // K-blocks of 32
  constexpr unsigned BBYTES = (unsigned)NW * KB * 1024u;  // staged B bytes
  static_assert(NT % NW == 0, "NW must divide NT");

  extern __shared__ f16_t lds_b[];

  int wavein = threadIdx.x >> 5;
  int lane   = threadIdx.x & 31;
  int ng  = blockIdx.x % NG;
  int mt0 = (blockIdx.x / NG) * 8 + wavein * 2;

  const f16_t* wgrp = wp + (size_t)ng * NW * KB * 512;  // contiguous B group

  // ---- stage B group into LDS via Tensor Data Mover (wave 0 only) ----
  if (wavein == 0) {
    unsigned lds_off = (unsigned)(size_t)lds_b;        // LDS byte offset
    unsigned long long ga = (unsigned long long)(size_t)wgrp;
    unsigned n8 = BBYTES >> 3;                         // 8-byte units
    u32x4 g0; i32x8 g1; i32x4 g2 = {}, g3 = {};
    g0[0] = 1u;                                        // count=1 (valid D#)
    g0[1] = lds_off;                                   // lds_addr
    g0[2] = (unsigned)ga;                              // global_addr[31:0]
    g0[3] = (unsigned)((ga >> 32) & 0x1FFFFFFu) | (2u << 30);  // ga[56:32]|type=2
    g1[0] = (int)(3u << 16);                           // data_size=8B, no flags
    g1[1] = (int)((n8 & 0xFFFFu) << 16);               // tensor_dim0[15:0]
    g1[2] = (int)((n8 >> 16) | (1u << 16));            // tensor_dim0[31:16], tensor_dim1=1
    g1[3] = (int)((n8 & 0xFFFFu) << 16);               // tile_dim0 = n8
    g1[4] = 1;                                         // tile_dim1=1, tile_dim2=0
    g1[5] = (int)n8;                                   // tensor_dim0_stride lo
    g1[6] = (int)((n8 & 0xFFFFu) << 16);               // stride hi=0, dim1_stride lo
    g1[7] = (int)(n8 >> 16);                           // dim1_stride hi
#if defined(__clang_major__) && (__clang_major__ >= 23)
    __builtin_amdgcn_tensor_load_to_lds(g0, g1, g2, g3, (i32x8){}, 0);
#else
    __builtin_amdgcn_tensor_load_to_lds(g0, g1, g2, g3, 0);
#endif
    __builtin_amdgcn_s_wait_tensorcnt(0);
  }
  __syncthreads();

  int m  = lane & 15;
  int kb = (lane < 16) ? 0 : 8;
  int w0[2], h[2], b[2], wpix[2];
#pragma unroll
  for (int i = 0; i < 2; ++i) {
    int mt = mt0 + i;
    w0[i]   = (mt % (Wc / 16)) * 16;
    h[i]    = (mt / (Wc / 16)) % Hc;
    b[i]    = mt / ((Wc / 16) * Hc);
    wpix[i] = w0[i] + m;
  }

  v8f acc[2][NW];
#pragma unroll
  for (int i = 0; i < 2; ++i)
#pragma unroll
    for (int w = 0; w < NW; ++w) acc[i][w] = (v8f){};

  for (int kk = 0; kk < KB; ++kk) {
    int s  = kk / CB;
    int cb = (kk % CB) * 32;
    v16h a[2];
#pragma unroll
    for (int i = 0; i < 2; ++i) {
      int hh = (KS == 3) ? (h[i] + s / 3 - 1) : h[i];
      int ww = (KS == 3) ? (wpix[i] + s % 3 - 1) : wpix[i];
      a[i] = (v16h){};
      if (hh >= 0 && hh < Hc && ww >= 0 && ww < Wc) {
        const f16_t* p = in + ((size_t)((b[i] * Hc + hh) * Wc + ww)) * CIN + cb + kb;
        v8h lo = *(const v8h*)p;
        v8h hi = *(const v8h*)(p + 16);
#pragma unroll
        for (int j = 0; j < 8; ++j) { a[i][j] = lo[j]; a[i][8 + j] = hi[j]; }
      }
    }
#pragma unroll
    for (int w = 0; w < NW; ++w) {
      v16h bf = *(const v16h*)(lds_b + ((size_t)(w * KB + kk) * 32 + lane) * 16);
#pragma unroll
      for (int i = 0; i < 2; ++i)
        acc[i][w] = __builtin_amdgcn_wmma_f32_16x16x32_f16(false, a[i], false, bf,
                                                           (short)0, acc[i][w],
                                                           false, false);
    }
  }

  int mb = (lane < 16) ? 0 : 8;
#pragma unroll
  for (int w = 0; w < NW; ++w) {
    int n = (ng * NW + w) * 16 + m;
    float al = alpha[n], be = beta[n];
#pragma unroll
    for (int i = 0; i < 2; ++i) {
#pragma unroll
      for (int r = 0; r < 8; ++r) {
        float v = acc[i][w][r] * al + be;
        if (LRELU) v = (v >= 0.0f) ? v : 0.2f * v;
        int pix = (b[i] * Hc + h[i]) * Wc + w0[i] + mb + r;
        if (MASK) {
          if (idx[pix] == expert) out[(size_t)pix * COUT + n] = (f16_t)v;
        } else {
          out[(size_t)pix * COUT + n] = (f16_t)v;
        }
      }
    }
  }
}

// ---------------------------------------------------------------------------
// cls5 head: per-pixel 512->20, softmax, argmax -> idx; LDS-reduced density
// ---------------------------------------------------------------------------
__global__ __launch_bounds__(256) void k_cls5(const f16_t* __restrict__ h4,
                                              const float* __restrict__ w5,
                                              const float* __restrict__ b5,
                                              int* __restrict__ idx,
                                              float* __restrict__ proxy,
                                              float* __restrict__ dens) {
  __shared__ float lprox[Ec];
  __shared__ float lcnt[Ec];
  int t = threadIdx.x;
  if (t < Ec) { lprox[t] = 0.0f; lcnt[t] = 0.0f; }
  __syncthreads();

  int pix = blockIdx.x * blockDim.x + t;
  float s[Ec];
#pragma unroll
  for (int e = 0; e < Ec; ++e) s[e] = b5[e];
  const f16_t* hp = h4 + (size_t)pix * 512;
  for (int c = 0; c < 512; ++c) {
    float hv = (float)hp[c];
#pragma unroll
    for (int e = 0; e < Ec; ++e) s[e] += hv * w5[e * 512 + c];
  }
  float mx = s[0];
  int am = 0;
#pragma unroll
  for (int e = 1; e < Ec; ++e)
    if (s[e] > mx) { mx = s[e]; am = e; }
  float den = 0.0f;
#pragma unroll
  for (int e = 0; e < Ec; ++e) { s[e] = __expf(s[e] - mx); den += s[e]; }
  float inv = 1.0f / den;
  idx[pix] = am;
#pragma unroll
  for (int e = 0; e < Ec; ++e) atomicAdd(&lprox[e], s[e] * inv);
  atomicAdd(&lcnt[am], 1.0f);
  __syncthreads();
  int b = (blockIdx.x * (int)blockDim.x) / HWc;  // block never straddles images
  if (t < Ec) {
    atomicAdd(&proxy[b * Ec + t], lprox[t]);
    atomicAdd(&dens[b * Ec + t], lcnt[t]);
  }
}

// ---------------------------------------------------------------------------
// emb (f16 NHWC 128) + noise (f32 NCHW 128) -> f16 NHWC 128
// ---------------------------------------------------------------------------
__global__ __launch_bounds__(256) void k_addnoise(const f16_t* __restrict__ emb,
                                                  const float* __restrict__ noise,
                                                  f16_t* __restrict__ t16) {
  size_t i = (size_t)blockIdx.x * blockDim.x + threadIdx.x;
  if (i >= (size_t)PIXc * 128) return;
  int c = (int)(i & 127);
  size_t pix = i >> 7;
  int b = (int)(pix / HWc);
  int hw = (int)(pix % HWc);
  float v = (float)emb[i] + noise[((size_t)(b * 128 + c)) * HWc + hw];
  t16[i] = (f16_t)v;
}

// ---------------------------------------------------------------------------
// Decoder tail: 32 -> 3 (1x1, bias), f16 NHWC in, f32 NCHW out.
// ---------------------------------------------------------------------------
__global__ __launch_bounds__(256) void k_dec3(const f16_t* __restrict__ d2,
                                              const float* __restrict__ w3,
                                              const float* __restrict__ b3,
                                              float* __restrict__ out) {
  int pix = blockIdx.x * blockDim.x + threadIdx.x;
  if (pix >= PIXc) return;
  int b = pix / HWc, hw = pix % HWc;
  const f16_t* p = d2 + (size_t)pix * 32;
#pragma unroll
  for (int o = 0; o < 3; ++o) {
    float s = b3[o];
    for (int c = 0; c < 32; ++c) s += (float)p[c] * w3[o * 32 + c];
    out[((size_t)(b * 3 + o)) * HWc + hw] = s;
  }
}

// ---------------------------------------------------------------------------
// lb = sum_{b,e} proxy_sum*count / HW^2   (algebraically == reference)
// ---------------------------------------------------------------------------
__global__ void k_lb(const float* __restrict__ proxy, const float* __restrict__ dens,
                     float* __restrict__ out) {
  if (threadIdx.x == 0 && blockIdx.x == 0) {
    float s = 0.0f;
    for (int i = 0; i < Bc * Ec; ++i) s += proxy[i] * dens[i];
    out[(size_t)Bc * 3 * HWc] = s / ((float)HWc * (float)HWc);
  }
}

// ---------------------------------------------------------------------------
// host-side launch helpers
// ---------------------------------------------------------------------------
template <int CIN, int COUT, int KS, int NW, bool LRELU, bool MASK>
static void conv_launch(const f16_t* in, const f16_t* wp, const float* al,
                        const float* be, f16_t* out, const int* idx, int e,
                        hipStream_t s) {
  constexpr int NG = (COUT / 16) / NW;
  constexpr int KB = KS * KS * (CIN / 32);
  constexpr unsigned SH = (unsigned)NW * KB * 1024u;
  k_conv_wmma<CIN, COUT, KS, NW, LRELU, MASK>
      <<<(MTc / 8) * NG, 128, SH, s>>>(in, wp, al, be, out, idx, e);
}

static void pack_launch(const float* w, f16_t* pk, int Cout, int Cin, int taps,
                        hipStream_t s) {
  int K = (Cin == 3) ? 32 : taps * Cin;
  int total = (Cout / 16) * (K / 32) * 512;
  k_pack<<<(total + 255) / 256, 256, 0, s>>>(w, pk, Cout, Cin, taps);
}

static void aff_launch(const float* bias, const float* g, const float* bb,
                       const float* mn, const float* vr, float* al, float* be,
                       int C, int bn, hipStream_t s) {
  k_affine<<<(C + 255) / 256, 256, 0, s>>>(bias, g, bb, mn, vr, al, be, C, bn);
}

extern "C" void kernel_launch(void* const* d_in, const int* in_sizes, int n_in,
                              void* d_out, int out_size, void* d_ws, size_t ws_size,
                              hipStream_t stream) {
  const float* x      = (const float*)d_in[0];
  const float* noise  = (const float*)d_in[1];
  const float* cls_w1 = (const float*)d_in[2];
  const float* cls_b1 = (const float*)d_in[3];
  const float* cls_w2 = (const float*)d_in[4];
  const float* cls_b2 = (const float*)d_in[5];
  const float* cls_w3 = (const float*)d_in[6];
  const float* cls_b3 = (const float*)d_in[7];
  const float* cls_w4 = (const float*)d_in[8];
  const float* cls_b4 = (const float*)d_in[9];
  const float* cls_w5 = (const float*)d_in[10];
  const float* cls_b5 = (const float*)d_in[11];
  const float* bn1_g = (const float*)d_in[12];
  const float* bn1_b = (const float*)d_in[13];
  const float* bn1_m = (const float*)d_in[14];
  const float* bn1_v = (const float*)d_in[15];
  const float* bn2_g = (const float*)d_in[16];
  const float* bn2_b = (const float*)d_in[17];
  const float* bn2_m = (const float*)d_in[18];
  const float* bn2_v = (const float*)d_in[19];
  const float* bn3_g = (const float*)d_in[20];
  const float* bn3_b = (const float*)d_in[21];
  const float* bn3_m = (const float*)d_in[22];
  const float* bn3_v = (const float*)d_in[23];
  const float* enc_w1 = (const float*)d_in[24];
  const float* enc_b1 = (const float*)d_in[25];
  const float* enc_w2 = (const float*)d_in[26];
  const float* enc_b2 = (const float*)d_in[27];
  const float* enc_w3 = (const float*)d_in[28];
  const float* enc_b3 = (const float*)d_in[29];
  const float* enc_w4 = (const float*)d_in[30];
  const float* enc_b4 = (const float*)d_in[31];
  const float* dec_w1 = (const float*)d_in[32];
  const float* dec_w2 = (const float*)d_in[33];
  const float* dec_w3 = (const float*)d_in[34];
  const float* dec_b3 = (const float*)d_in[35];
  float* out = (float*)d_out;

  // ---- workspace carve ----
  size_t off = 0;
  auto alloc = [&](size_t bytes) -> char* {
    char* p = (char*)d_ws + off;
    off += (bytes + 255) & ~(size_t)255;
    return p;
  };
  f16_t* x16    = (f16_t*)alloc((size_t)PIXc * 3 * sizeof(f16_t));
  f16_t* bufA   = (f16_t*)alloc((size_t)PIXc * 256 * sizeof(f16_t));
  f16_t* bufB   = (f16_t*)alloc((size_t)PIXc * 512 * sizeof(f16_t));
  f16_t* emb    = (f16_t*)alloc((size_t)PIXc * 128 * sizeof(f16_t));
  f16_t* packed = (f16_t*)alloc((size_t)1 << 20);
  int*   idx    = (int*)alloc((size_t)PIXc * sizeof(int));
  float* alpha  = (float*)alloc(512 * sizeof(float));
  float* beta   = (float*)alloc(512 * sizeof(float));
  float* proxy  = (float*)alloc((size_t)2 * Bc * Ec * sizeof(float));
  float* dens   = proxy + Bc * Ec;
  if (off > ws_size) return;  // insufficient scratch: do nothing (deterministic)
  (void)in_sizes; (void)n_in; (void)out_size;

  hipMemsetAsync(proxy, 0, (size_t)2 * Bc * Ec * sizeof(float), stream);

  k_x_to_f16<<<(Bc * 3 * HWc + 255) / 256, 256, 0, stream>>>(x, x16);

  // ---- classification head ----
  aff_launch(cls_b1, nullptr, nullptr, nullptr, nullptr, alpha, beta, 64, 0, stream);
  pack_launch(cls_w1, packed, 64, 3, 9, stream);
  k_conv_c3<64, true><<<(MTc * 4) / 4, 128, 0, stream>>>(x16, packed, alpha, beta, bufA);

  aff_launch(cls_b2, bn1_g, bn1_b, bn1_m, bn1_v, alpha, beta, 128, 1, stream);
  pack_launch(cls_w2, packed, 128, 64, 9, stream);
  conv_launch<64, 128, 3, 4, true, false>(bufA, packed, alpha, beta, bufB, nullptr, 0, stream);

  aff_launch(cls_b3, bn2_g, bn2_b, bn2_m, bn2_v, alpha, beta, 256, 1, stream);
  pack_launch(cls_w3, packed, 256, 128, 9, stream);
  conv_launch<128, 256, 3, 4, true, false>(bufB, packed, alpha, beta, bufA, nullptr, 0, stream);

  aff_launch(cls_b4, bn3_g, bn3_b, bn3_m, bn3_v, alpha, beta, 512, 1, stream);
  pack_launch(cls_w4, packed, 512, 256, 1, stream);
  conv_launch<256, 512, 1, 4, true, false>(bufA, packed, alpha, beta, bufB, nullptr, 0, stream);

  k_cls5<<<PIXc / 256, 256, 0, stream>>>(bufB, cls_w5, cls_b5, idx, proxy, dens);

  // ---- 20 expert encoders; last conv masked-scatters into emb ----
  for (int e = 0; e < Ec; ++e) {
    aff_launch(enc_b1 + e * 32, nullptr, nullptr, nullptr, nullptr, alpha, beta, 32, 0, stream);
    pack_launch(enc_w1 + (size_t)e * 32 * 3 * 9, packed, 32, 3, 9, stream);
    k_conv_c3<32, true><<<(MTc * 2) / 4, 128, 0, stream>>>(x16, packed, alpha, beta, bufA);

    aff_launch(enc_b2 + e * 64, nullptr, nullptr, nullptr, nullptr, alpha, beta, 64, 0, stream);
    pack_launch(enc_w2 + (size_t)e * 64 * 32 * 9, packed, 64, 32, 9, stream);
    conv_launch<32, 64, 3, 4, true, false>(bufA, packed, alpha, beta, bufB, nullptr, 0, stream);

    aff_launch(enc_b3 + e * 128, nullptr, nullptr, nullptr, nullptr, alpha, beta, 128, 0, stream);
    pack_launch(enc_w3 + (size_t)e * 128 * 64 * 9, packed, 128, 64, 9, stream);
    conv_launch<64, 128, 3, 4, true, false>(bufB, packed, alpha, beta, bufA, nullptr, 0, stream);

    aff_launch(enc_b4 + e * 128, nullptr, nullptr, nullptr, nullptr, alpha, beta, 128, 0, stream);
    pack_launch(enc_w4 + (size_t)e * 128 * 128 * 9, packed, 128, 128, 9, stream);
    conv_launch<128, 128, 3, 4, false, true>(bufA, packed, alpha, beta, emb, idx, e, stream);
  }

  // ---- decoder: (emb + noise) -> 64 -> 32 -> 3 (no activations) ----
  k_addnoise<<<(int)(((size_t)PIXc * 128 + 255) / 256), 256, 0, stream>>>(emb, noise, bufA);

  aff_launch(nullptr, nullptr, nullptr, nullptr, nullptr, alpha, beta, 64, 0, stream);
  pack_launch(dec_w1, packed, 64, 128, 1, stream);
  conv_launch<128, 64, 1, 4, false, false>(bufA, packed, alpha, beta, bufB, nullptr, 0, stream);

  aff_launch(nullptr, nullptr, nullptr, nullptr, nullptr, alpha, beta, 32, 0, stream);
  pack_launch(dec_w2, packed, 32, 64, 1, stream);
  conv_launch<64, 32, 1, 2, false, false>(bufB, packed, alpha, beta, bufA, nullptr, 0, stream);

  k_dec3<<<PIXc / 256, 256, 0, stream>>>(bufA, dec_w3, dec_b3, out);

  k_lb<<<1, 32, 0, stream>>>(proxy, dens, out);
}
